// MambaBlock_58342835749247
// MI455X (gfx1250) — compile-verified, matrix operand order checked by reading, not run on hardware
//
#include <hip/hip_runtime.h>
#include <hip/hip_bf16.h>

// ---------------------------------------------------------------------------
// Mamba block for MI455X (gfx1250).
//  - Operands of the two big GEMMs are pre-converted to bf16 (one streaming
//    pass each; v_cvt_pk_bf16_f32).
//  - Big GEMMs: LDS-tiled WMMA, 128x128x32 block tile, 8 waves x (2x4) 16x16
//    accumulators. Tiles are DMA'd into LDS by the Tensor Data Mover
//    (tensor_load_to_lds + s_wait_tensorcnt), double-buffered; the steady
//    state inner loop is ds_load_b128 + v_wmma_f32_16x16x32_bf16 only.
//  - Small GEMMs (N=96 / K=64), conv+SiLU, selective scan are as before.
// ---------------------------------------------------------------------------

#define DIM      1024
#define D_STATE  16
#define D_CONV   4
#define D_INNER  2048
#define DT_RANK  64
#define BATCH    2
#define SEQ      2048
#define ROWS     (BATCH * SEQ)          // 4096

typedef __attribute__((ext_vector_type(16))) __bf16        v16bf;
typedef __attribute__((ext_vector_type(8)))  float         v8f;
typedef __attribute__((ext_vector_type(8)))  unsigned int  v8u;
typedef __attribute__((ext_vector_type(2)))  float         f32x2;
typedef __attribute__((ext_vector_type(2)))  __bf16        bf16x2;
typedef __attribute__((ext_vector_type(4)))  unsigned int  u32x4;
typedef __attribute__((ext_vector_type(4)))  int           i32x4;
typedef __attribute__((ext_vector_type(8)))  int           i32x8;

// ---------------------------------------------------------------------------
// bf16 pack helpers (lower to v_cvt_pk_bf16_f32)
// ---------------------------------------------------------------------------
__device__ __forceinline__ unsigned int cvt_pk_bf16(float a, float b) {
  f32x2 f = {a, b};
  return __builtin_bit_cast(unsigned int, __builtin_convertvector(f, bf16x2));
}
__device__ __forceinline__ uint2 pack4_bf16(float4 v) {
  uint2 r;
  r.x = cvt_pk_bf16(v.x, v.y);
  r.y = cvt_pk_bf16(v.z, v.w);
  return r;
}

// ---------------------------------------------------------------------------
// WMMA fragment from LDS (ISA 05_wmma §7.12.2, 16-bit 16x32 MxK layout):
// lane&15 selects row, hi = 8*(lane>=16); 16B-contiguous pairs -> 2x b128.
// ---------------------------------------------------------------------------
__device__ __forceinline__ v16bf ld_frag_lds(const __bf16* base, int stride, int lane) {
  const int m  = lane & 15;
  const int hi = (lane >> 4) * 8;
  const __bf16* p = base + m * stride + hi;
  v8u u;
#pragma unroll
  for (int v = 0; v < 4; ++v) {
    u[v]     = *(const unsigned int*)(p + 2 * v);        // K = hi+2v, hi+2v+1
    u[4 + v] = *(const unsigned int*)(p + 16 + 2 * v);   // K = 16+hi+2v, ...
  }
  return __builtin_bit_cast(v16bf, u);
}

// Fragment straight from global f32 (small GEMM kernel).
__device__ __forceinline__ v16bf ld_frag_f32(const float* __restrict__ base,
                                             int ld, int row0, int k0, int lane) {
  const int r  = row0 + (lane & 15);
  const int hi = (lane >> 4) * 8;
  const float* p = base + (size_t)r * ld + k0 + hi;
  v16bf f;
#pragma unroll
  for (int v = 0; v < 4; ++v) {
    f[2 * v]         = (__bf16)p[2 * v];
    f[2 * v + 1]     = (__bf16)p[2 * v + 1];
    f[8 + 2 * v]     = (__bf16)p[16 + 2 * v];
    f[8 + 2 * v + 1] = (__bf16)p[16 + 2 * v + 1];
  }
  return f;
}

// ---------------------------------------------------------------------------
// Tensor Data Mover: 2D bf16 tile (rows x 32 elems) global -> LDS.
// D# per CDNA5 ISA:
//   g0: count=1 | lds_addr | global_addr | type=2
//   g1: data_size=2B, pad_enable, pad_interval=16 DWORDs (64B = one row),
//       pad_amount=4 DWORDs (16B) -> LDS row stride 80B = 40 bf16 elems,
//       tensor_dim0/1, tile_dim0=32, tile_dim1=rows, tensor_dim0_stride.
// ---------------------------------------------------------------------------
__device__ __forceinline__ void tdm_load_tile_bf16(const __bf16* gptr,
                                                   unsigned int lds_off,
                                                   int rows, int stride_elems) {
  const unsigned long long ga = (unsigned long long)gptr;
  u32x4 g0;
  g0.x = 1u;                                          // count=1, user mode
  g0.y = lds_off;                                     // LDS byte address
  g0.z = (unsigned int)(ga & 0xffffffffu);            // global_addr[31:0]
  g0.w = (unsigned int)((ga >> 32) & 0x1ffffffu)      // global_addr[56:32]
         | (2u << 30);                                // type=2 ("image")
  i32x8 g1;
  g1[0] = (1 << 16)                                   // data_size: 2 bytes
        | (1 << 20)                                   // pad_enable
        | (3 << 22)                                   // pad_interval: 16 DWORDs
        | (3 << 25);                                  // pad_amount: 4 DWORDs
  const int td0 = stride_elems;                       // tensor_dim0
  const int td1 = 1 << 20;                            // tensor_dim1 (no OOB)
  g1[1] = (td0 & 0xffff) << 16;                       // tensor_dim0[15:0]
  g1[2] = ((td0 >> 16) & 0xffff) | ((td1 & 0xffff) << 16);
  g1[3] = ((td1 >> 16) & 0xffff) | (32 << 16);        // tile_dim0 = 32
  g1[4] = rows & 0xffff;                              // tile_dim1
  g1[5] = stride_elems;                               // tensor_dim0_stride[31:0]
  g1[6] = 0;
  g1[7] = 0;
  const i32x4 z4 = {};
#if defined(__clang_major__) && (__clang_major__ >= 23)
  const i32x8 z8 = {};
  __builtin_amdgcn_tensor_load_to_lds(g0, g1, z4, z4, z8, 0);
#else
  __builtin_amdgcn_tensor_load_to_lds(g0, g1, z4, z4, 0);
#endif
}

// ---------------------------------------------------------------------------
// Tiled WMMA GEMM (bf16 in, f32 out): C(M,N) = A(M,K) * W(N,K)^T.
// grid = (N/128, M/128), block = 256 (8 waves). TDM-fed double-buffered LDS.
// ---------------------------------------------------------------------------
#define BM 128
#define BN 128
#define BK 32
#define LDSS 40           // bf16 elems per LDS row (32 + 8 pad = 80B)
#define TILE_ELEMS (BM * LDSS)

__global__ void __launch_bounds__(256)
wmma_gemm_tdm(const __bf16* __restrict__ A, int lda,
              const __bf16* __restrict__ W, int ldw,
              float* __restrict__ C, int ldc, int K) {
  __shared__ __bf16 As[2][TILE_ELEMS];
  __shared__ __bf16 Bs[2][TILE_ELEMS];

  const int tid  = threadIdx.x;
  const int lane = tid & 31;
  const int wave = tid >> 5;
  const int wm   = (wave >> 1) * 32;     // wave M offset inside block tile
  const int wn   = (wave & 1) * 64;      // wave N offset inside block tile
  const int blockM = blockIdx.y * BM;
  const int blockN = blockIdx.x * BN;

  // LDS byte offsets of the staging buffers (generic LDS addr low 32 bits).
  const unsigned int asOff0 = (unsigned int)(size_t)&As[0][0];
  const unsigned int asOff1 = (unsigned int)(size_t)&As[1][0];
  const unsigned int bsOff0 = (unsigned int)(size_t)&Bs[0][0];
  const unsigned int bsOff1 = (unsigned int)(size_t)&Bs[1][0];

  v8f acc[2][4];
#pragma unroll
  for (int i = 0; i < 2; ++i)
#pragma unroll
    for (int j = 0; j < 4; ++j) acc[i][j] = v8f{};

  auto issue = [&](int k0, int stage) {
    tdm_load_tile_bf16(A + (size_t)blockM * lda + k0,
                       stage ? asOff1 : asOff0, BM, lda);
    tdm_load_tile_bf16(W + (size_t)blockN * ldw + k0,
                       stage ? bsOff1 : bsOff0, BN, ldw);
  };

  if (tid < 32) issue(0, 0);             // prologue: stage 0

  int cur = 0;
  for (int k0 = 0; k0 < K; k0 += BK) {
    if (tid < 32) __builtin_amdgcn_s_wait_tensorcnt(0);
    __syncthreads();                     // buffer `cur` ready for everyone
    if (k0 + BK < K && tid < 32) issue(k0 + BK, cur ^ 1);  // prefetch

    v16bf af[2], bfr[4];
#pragma unroll
    for (int i = 0; i < 2; ++i)
      af[i] = ld_frag_lds(&As[cur][(wm + i * 16) * LDSS], LDSS, lane);
#pragma unroll
    for (int j = 0; j < 4; ++j)
      bfr[j] = ld_frag_lds(&Bs[cur][(wn + j * 16) * LDSS], LDSS, lane);

#pragma unroll
    for (int i = 0; i < 2; ++i)
#pragma unroll
      for (int j = 0; j < 4; ++j)
        acc[i][j] = __builtin_amdgcn_wmma_f32_16x16x32_bf16(
            false, af[i], false, bfr[j], (short)0, acc[i][j], false, false);

    cur ^= 1;
  }

  // D layout: lanes 0-15 -> N=lane, M=r; lanes 16-31 -> N=lane-16, M=r+8.
  const int n  = lane & 15;
  const int mb = (lane >> 4) * 8;
#pragma unroll
  for (int i = 0; i < 2; ++i) {
    const int row0 = blockM + wm + i * 16 + mb;
#pragma unroll
    for (int j = 0; j < 4; ++j) {
      const int col = blockN + wn + j * 16 + n;
#pragma unroll
      for (int r = 0; r < 8; ++r)
        C[(size_t)(row0 + r) * ldc + col] = acc[i][j][r];
    }
  }
}

// ---------------------------------------------------------------------------
// f32 -> bf16 streaming conversion (float4 in, 2 dwords out).
// ---------------------------------------------------------------------------
__global__ void f32_to_bf16_kernel(const float* __restrict__ src,
                                   __bf16* __restrict__ dst) {
  const int i = blockIdx.x * blockDim.x + threadIdx.x;
  const float4 v = ((const float4*)src)[i];
  ((uint2*)dst)[i] = pack4_bf16(v);
}

// ---------------------------------------------------------------------------
// Small GEMM: one 16x16 tile per wave32 (N=96 and K=64 projections).
// EPI: 0 = plain, 1 = bias + softplus.
// ---------------------------------------------------------------------------
template <int EPI>
__global__ void wmma_gemm_small(const float* __restrict__ A, int lda,
                                const float* __restrict__ W, int ldw,
                                const float* __restrict__ bias,
                                float* __restrict__ C, int ldc, int K) {
  const int lane = threadIdx.x & 31;
  const int row0 = blockIdx.y * 16;
  const int col0 = blockIdx.x * 16;

  v8f acc = {};
  for (int k0 = 0; k0 < K; k0 += 32) {
    v16bf a = ld_frag_f32(A, lda, row0, k0, lane);
    v16bf b = ld_frag_f32(W, ldw, col0, k0, lane);
    acc = __builtin_amdgcn_wmma_f32_16x16x32_bf16(
        false, a, false, b, (short)0, acc, false, false);
  }

  const int n  = lane & 15;
  const int mb = (lane >> 4) * 8;
  float bv = 0.0f;
  if (EPI == 1) bv = bias[col0 + n];
#pragma unroll
  for (int r = 0; r < 8; ++r) {
    float v = acc[r];
    if (EPI == 1) {                       // softplus(v + b_dt)
      v += bv;
      v = (v > 20.0f) ? v : log1pf(__expf(v));
    }
    C[(size_t)(row0 + mb + r) * ldc + col0 + n] = v;
  }
}

// ---------------------------------------------------------------------------
// Causal depthwise conv (K=4) over L, + bias + SiLU.
// ---------------------------------------------------------------------------
__global__ void conv_silu_kernel(const float* __restrict__ xz,
                                 const float* __restrict__ conv_w,
                                 const float* __restrict__ conv_b,
                                 float* __restrict__ xc_act) {
  const int idx = blockIdx.x * blockDim.x + threadIdx.x;     // ROWS*D_INNER
  const int d   = idx & (D_INNER - 1);
  const int row = idx >> 11;
  const int l   = row & (SEQ - 1);
  const int b   = row >> 11;

  float acc = conv_b[d];
#pragma unroll
  for (int k = 0; k < D_CONV; ++k) {
    const int ll = l + k - (D_CONV - 1);
    if (ll >= 0)
      acc += xz[(size_t)(b * SEQ + ll) * (2 * D_INNER) + d] * conv_w[d * D_CONV + k];
  }
  acc = acc / (1.0f + __expf(-acc));                          // SiLU
  xc_act[(size_t)row * D_INNER + d] = acc;
}

// ---------------------------------------------------------------------------
// Selective scan: one thread per (b, d), 16 states in registers, fused
// y *= silu(z); y written directly as bf16 for the output GEMM.
// ---------------------------------------------------------------------------
__global__ void scan_kernel(const float* __restrict__ xc,
                            const float* __restrict__ delta,
                            const float* __restrict__ x_dbl,  // (ROWS, 96)
                            const float* __restrict__ A_log,
                            const float* __restrict__ Dvec,
                            const float* __restrict__ xz,
                            __bf16* __restrict__ y_bf) {
  const int idx = blockIdx.x * blockDim.x + threadIdx.x;      // (b, d)
  const int b = idx >> 11;
  const int d = idx & (D_INNER - 1);

  float Aneg[D_STATE];
#pragma unroll
  for (int n = 0; n < D_STATE; ++n) Aneg[n] = -__expf(A_log[d * D_STATE + n]);
  const float Dd = Dvec[d];

  float h[D_STATE];
#pragma unroll
  for (int n = 0; n < D_STATE; ++n) h[n] = 0.0f;

  for (int l = 0; l < SEQ; ++l) {
    const size_t row = (size_t)b * SEQ + l;
    const float dlt = delta[row * D_INNER + d];
    const float u   = xc[row * D_INNER + d];
    const float du  = dlt * u;
    const float4* bc4 = (const float4*)(x_dbl + row * (DT_RANK + 2 * D_STATE) + DT_RANK);
    float Bv[D_STATE], Cv[D_STATE];
#pragma unroll
    for (int q = 0; q < 4; ++q) {
      const float4 bq = bc4[q];
      const float4 cq = bc4[4 + q];
      Bv[4 * q] = bq.x; Bv[4 * q + 1] = bq.y; Bv[4 * q + 2] = bq.z; Bv[4 * q + 3] = bq.w;
      Cv[4 * q] = cq.x; Cv[4 * q + 1] = cq.y; Cv[4 * q + 2] = cq.z; Cv[4 * q + 3] = cq.w;
    }
    float acc = 0.0f;
#pragma unroll
    for (int n = 0; n < D_STATE; ++n) {
      const float dA = __expf(dlt * Aneg[n]);
      h[n] = dA * h[n] + du * Bv[n];
      acc += h[n] * Cv[n];
    }
    float yv = acc + u * Dd;
    const float z = xz[row * (2 * D_INNER) + D_INNER + d];
    yv *= z / (1.0f + __expf(-z));                 // * silu(z)
    y_bf[row * D_INNER + d] = (__bf16)yv;
  }
}

// ---------------------------------------------------------------------------
// Workspace layout (bytes)
// ---------------------------------------------------------------------------
#define XZ_OFF    ((size_t)0)                                    // 4096*4096 f32
#define XC_OFF    (XZ_OFF    + (size_t)ROWS * 2 * D_INNER * 4)   // 4096*2048 f32
#define XDBL_OFF  (XC_OFF    + (size_t)ROWS * D_INNER * 4)       // 4096*96   f32
#define DELTA_OFF (XDBL_OFF  + (size_t)ROWS * 96 * 4)            // 4096*2048 f32
#define YBF_OFF   (DELTA_OFF + (size_t)ROWS * D_INNER * 4)       // 4096*2048 bf16
#define XBF_OFF   (YBF_OFF   + (size_t)ROWS * D_INNER * 2)       // 4096*1024 bf16
#define WINBF_OFF (XBF_OFF   + (size_t)ROWS * DIM * 2)           // 4096*1024 bf16
#define WOUTBF_OFF (WINBF_OFF + (size_t)(2 * D_INNER) * DIM * 2) // 1024*2048 bf16

extern "C" void kernel_launch(void* const* d_in, const int* in_sizes, int n_in,
                              void* d_out, int out_size, void* d_ws, size_t ws_size,
                              hipStream_t stream) {
  (void)in_sizes; (void)n_in; (void)out_size; (void)ws_size;

  const float* x       = (const float*)d_in[0];   // (2,2048,1024)
  const float* W_in    = (const float*)d_in[1];   // (4096,1024)
  const float* conv_w  = (const float*)d_in[2];   // (2048,4)
  const float* conv_b  = (const float*)d_in[3];   // (2048)
  const float* W_xproj = (const float*)d_in[4];   // (96,2048)
  const float* W_dt    = (const float*)d_in[5];   // (2048,64)
  const float* b_dt    = (const float*)d_in[6];   // (2048)
  const float* A_log   = (const float*)d_in[7];   // (2048,16)
  const float* Dvec    = (const float*)d_in[8];   // (2048)
  const float* W_out   = (const float*)d_in[9];   // (1024,2048)
  float* out = (float*)d_out;                     // (2,2048,1024)

  char* ws = (char*)d_ws;
  float*  xz      = (float*)(ws + XZ_OFF);
  float*  xc      = (float*)(ws + XC_OFF);
  float*  x_dbl   = (float*)(ws + XDBL_OFF);
  float*  delta   = (float*)(ws + DELTA_OFF);
  __bf16* y_bf    = (__bf16*)(ws + YBF_OFF);
  __bf16* x_bf    = (__bf16*)(ws + XBF_OFF);
  __bf16* Win_bf  = (__bf16*)(ws + WINBF_OFF);
  __bf16* Wout_bf = (__bf16*)(ws + WOUTBF_OFF);

  // 0) bf16 copies of the big-GEMM operands
  f32_to_bf16_kernel<<<(ROWS * DIM / 4) / 256, 256, 0, stream>>>(x, x_bf);
  f32_to_bf16_kernel<<<(2 * D_INNER * DIM / 4) / 256, 256, 0, stream>>>(W_in, Win_bf);
  f32_to_bf16_kernel<<<(DIM * D_INNER / 4) / 256, 256, 0, stream>>>(W_out, Wout_bf);

  // 1) xz = x @ W_in^T                      (4096 x 4096 x 1024)
  wmma_gemm_tdm<<<dim3((2 * D_INNER) / BN, ROWS / BM), 256, 0, stream>>>(
      x_bf, DIM, Win_bf, DIM, xz, 2 * D_INNER, DIM);

  // 2) xc = silu(causal_dwconv(xz[:, :2048]) + conv_b)
  conv_silu_kernel<<<(ROWS * D_INNER) / 256, 256, 0, stream>>>(xz, conv_w, conv_b, xc);

  // 3) x_dbl = xc @ W_xproj^T               (4096 x 96 x 2048)
  wmma_gemm_small<0><<<dim3(96 / 16, ROWS / 16), 32, 0, stream>>>(
      xc, D_INNER, W_xproj, D_INNER, nullptr, x_dbl, 96, D_INNER);

  // 4) delta = softplus(x_dbl[:, :64] @ W_dt^T + b_dt)   (4096 x 2048 x 64)
  wmma_gemm_small<1><<<dim3(D_INNER / 16, ROWS / 16), 32, 0, stream>>>(
      x_dbl, 96, W_dt, DT_RANK, b_dt, delta, D_INNER, DT_RANK);

  // 5) selective scan + y *= silu(z), bf16 output
  scan_kernel<<<ROWS / 256, 256, 0, stream>>>(xc, delta, x_dbl, A_log, Dvec, xz, y_bf);

  // 6) out = y @ W_out^T                    (4096 x 1024 x 2048)
  wmma_gemm_tdm<<<dim3(DIM / BN, ROWS / BM), 256, 0, stream>>>(
      y_bf, D_INNER, Wout_bf, D_INNER, out, DIM, D_INNER);
}
